// Model_10496900071610
// MI455X (gfx1250) — matrix operand order repaired and checked by reading, not compiled, not run on hardware
//
#include <hip/hip_runtime.h>
#include <cstdint>
#include <cstddef>

#define ND_N    20000
#define NP_N    8000
#define E_N     640000
#define L_N     200000
#define H_N     256

typedef __attribute__((ext_vector_type(16))) __bf16 v16bf;
typedef __attribute__((ext_vector_type(8)))  __bf16 v8bf;
typedef __attribute__((ext_vector_type(4)))  __bf16 v4bf;
typedef __attribute__((ext_vector_type(8)))  float  v8f;

__device__ inline v8f v8f_zero() {
  v8f z;
#pragma unroll
  for (int i = 0; i < 8; ++i) z[i] = 0.0f;
  return z;
}

// ---------------------------------------------------------------------------
// B fragment: WT is the transposed weight, [256 x K] row-major in bf16.
// ISA 16-bit B layout (32x16): lanes 0-15: elems 0..7 -> K=0..7,
// elems 8..15 -> K=16..23 (N = lane); lanes 16-31: bases 8 and 24 (N = lane-16).
// ---------------------------------------------------------------------------
__device__ inline v16bf load_b_frag(const __bf16* __restrict__ wt, int K,
                                    int col0, int k0, int lane) {
  const int n  = lane & 15;
  const int kb = (lane >> 4) << 3;   // 0 or 8
  const __bf16* p = wt + (size_t)(col0 + n) * K + (k0 + kb);
  const v8bf lo = *reinterpret_cast<const v8bf*>(p);
  const v8bf hi = *reinterpret_cast<const v8bf*>(p + 16);
  v16bf b;
#pragma unroll
  for (int i = 0; i < 8; ++i) {
    b[i]     = lo[i];
    b[i + 8] = hi[i];
  }
  return b;
}

// ---------------------------------------------------------------------------
// A fragment from bf16 row-major [*, K]; strip = base of the 16-row tile.
// ISA 16-bit A layout (16x32): lanes 0-15: elems 0..7 -> K=0..7,
// elems 8..15 -> K=16..23; lanes 16-31: bases 8 and 24.
// Two contiguous 16B runs -> no converts in the inner loop.
// ---------------------------------------------------------------------------
__device__ inline v16bf load_a_frag(const __bf16* __restrict__ strip, int K,
                                    int k0, int lane) {
  const int m  = lane & 15;
  const int kb = (lane >> 4) << 3;   // 0 or 8
  const __bf16* p = strip + (size_t)m * K + (k0 + kb);
  const v8bf lo = *reinterpret_cast<const v8bf*>(p);        // K = kb .. kb+7
  const v8bf hi = *reinterpret_cast<const v8bf*>(p + 16);   // K = kb+16 .. kb+23
  v16bf a;
#pragma unroll
  for (int i = 0; i < 8; ++i) {
    a[i]     = lo[i];
    a[i + 8] = hi[i];
  }
  return a;
}

// ---------------------------------------------------------------------------
// small utility kernels
// ---------------------------------------------------------------------------
__global__ void zero_i32(int* __restrict__ p, int n) {
  const int i = blockIdx.x * blockDim.x + threadIdx.x;
  if (i < n) p[i] = 0;
}

__global__ void copy_i32(const int* __restrict__ src, int* __restrict__ dst, int n) {
  const int i = blockIdx.x * blockDim.x + threadIdx.x;
  if (i < n) dst[i] = src[i];
}

__global__ void f32_to_bf16(const float* __restrict__ x, __bf16* __restrict__ y, int n) {
  const int i = blockIdx.x * blockDim.x + threadIdx.x;
  if (i < n) y[i] = (__bf16)x[i];
}

// degree histogram: deg[key[e]] += 1
__global__ void hist_i32(const int* __restrict__ key, int* __restrict__ deg, int E) {
  const int e = blockIdx.x * blockDim.x + threadIdx.x;
  if (e < E) atomicAdd(&deg[key[e]], 1);
}

// single-block exclusive scan (Hillis-Steele in LDS, chunked with carry)
__global__ void scan_excl(const int* __restrict__ deg, int* __restrict__ offs, int M) {
  __shared__ int lds[1024];
  __shared__ int carry;
  const int tid = threadIdx.x;
  if (tid == 0) carry = 0;
  __syncthreads();
  for (int base = 0; base < M; base += 1024) {
    const int i = base + tid;
    const int v = (i < M) ? deg[i] : 0;
    lds[tid] = v;
    __syncthreads();
#pragma unroll
    for (int off = 1; off < 1024; off <<= 1) {
      const int t = (tid >= off) ? lds[tid - off] : 0;
      __syncthreads();
      lds[tid] += t;
      __syncthreads();
    }
    if (i < M) offs[i] = carry + (lds[tid] - v);   // exclusive
    __syncthreads();
    if (tid == 1023) carry += lds[1023];
    __syncthreads();
  }
  if (tid == 0) offs[M] = carry;
}

// place neighbor node-ids into CSR value array
__global__ void csr_place(const int* __restrict__ key, const int* __restrict__ val,
                          int* __restrict__ cursor, int* __restrict__ out_vals, int E) {
  const int e = blockIdx.x * blockDim.x + threadIdx.x;
  if (e < E) {
    const int pos = atomicAdd(&cursor[key[e]], 1);
    out_vals[pos] = val[e];
  }
}

// ---------------------------------------------------------------------------
// CSR gather-sum (bf16 in, fp32 accumulate, bf16 out):
//   agg[n, :] = sum over neighbors s of X[s, :]
// One wave per destination node.
//  F==256: lane owns 8 contiguous columns -> one b128 load per edge per lane.
//  F==128: lane owns 4 contiguous columns -> one b64  load per edge per lane.
// No atomics; 2x unrolled so multiple edge-row loads are in flight.
// ---------------------------------------------------------------------------
__global__ __launch_bounds__(256, 8)
void gather_sum_bf16(const __bf16* __restrict__ X, const int* __restrict__ offs,
                     const int* __restrict__ vals, __bf16* __restrict__ agg,
                     int M, int F) {
  const int wave = threadIdx.x >> 5;
  const int lane = threadIdx.x & 31;
  const int node = blockIdx.x * 8 + wave;
  if (node >= M) return;
  const int beg = offs[node];
  const int end = offs[node + 1];

  if (F == 256) {
    const int c0 = lane << 3;               // 8 contiguous columns per lane
    float a[8] = {0.f, 0.f, 0.f, 0.f, 0.f, 0.f, 0.f, 0.f};
    int i = beg;
    for (; i + 1 < end; i += 2) {
      const int s0 = vals[i];
      const int s1 = vals[i + 1];
      const v8bf u = *reinterpret_cast<const v8bf*>(X + (size_t)s0 * 256 + c0);
      const v8bf w = *reinterpret_cast<const v8bf*>(X + (size_t)s1 * 256 + c0);
#pragma unroll
      for (int j = 0; j < 8; ++j) { a[j] += (float)u[j]; a[j] += (float)w[j]; }
    }
    if (i < end) {
      const v8bf u = *reinterpret_cast<const v8bf*>(X + (size_t)vals[i] * 256 + c0);
#pragma unroll
      for (int j = 0; j < 8; ++j) a[j] += (float)u[j];
    }
    v8bf o;
#pragma unroll
    for (int j = 0; j < 8; ++j) o[j] = (__bf16)a[j];
    *reinterpret_cast<v8bf*>(agg + (size_t)node * 256 + c0) = o;
  } else {                                   // F == 128
    const int c0 = lane << 2;               // 4 contiguous columns per lane
    float a[4] = {0.f, 0.f, 0.f, 0.f};
    int i = beg;
    for (; i + 1 < end; i += 2) {
      const int s0 = vals[i];
      const int s1 = vals[i + 1];
      const v4bf u = *reinterpret_cast<const v4bf*>(X + (size_t)s0 * 128 + c0);
      const v4bf w = *reinterpret_cast<const v4bf*>(X + (size_t)s1 * 128 + c0);
#pragma unroll
      for (int j = 0; j < 4; ++j) { a[j] += (float)u[j]; a[j] += (float)w[j]; }
    }
    if (i < end) {
      const v4bf u = *reinterpret_cast<const v4bf*>(X + (size_t)vals[i] * 128 + c0);
#pragma unroll
      for (int j = 0; j < 4; ++j) a[j] += (float)u[j];
    }
    v4bf o;
#pragma unroll
    for (int j = 0; j < 4; ++j) o[j] = (__bf16)a[j];
    *reinterpret_cast<v4bf*>(agg + (size_t)node * 128 + c0) = o;
  }
}

// ---------------------------------------------------------------------------
// W [K, 256] fp32  ->  WT [256, K] bf16 (transpose + convert)
// ---------------------------------------------------------------------------
__global__ void convert_wT(const float* __restrict__ W, __bf16* __restrict__ WT,
                           int K, int N) {
  const int idx = blockIdx.x * blockDim.x + threadIdx.x;
  if (idx >= K * N) return;
  const int k = idx / N;
  const int n = idx - k * N;
  WT[(size_t)n * K + k] = (__bf16)W[idx];
}

// ---------------------------------------------------------------------------
// Fused SAGE layer GEMM (all-bf16 inputs, f32 accumulate):
//   Z[M,256] = A[M,K] @ WlT^T + X[M,K] @ WrT^T + bias  (opt tanh)
// Always writes bf16 activations Zb; optionally fp32 Zf (final layer).
// Wave: 16 rows x 64 cols (4 WMMA accumulators). Block: 8 waves = 128 rows.
// ---------------------------------------------------------------------------
__global__ __launch_bounds__(256, 4)
void sage_gemm_wmma(const __bf16* __restrict__ A, const __bf16* __restrict__ X,
                    const __bf16* __restrict__ WlT, const __bf16* __restrict__ WrT,
                    const float* __restrict__ bias,
                    __bf16* __restrict__ Zb, float* __restrict__ Zf,
                    int M, int K, int do_tanh) {
  const int lane = threadIdx.x & 31;
  const int wave = threadIdx.x >> 5;
  const int row0 = blockIdx.x * 128 + wave * 16;
  if (row0 >= M) return;                 // wave-uniform; EXEC stays all-ones
  const int col0 = blockIdx.y * 64;

  v8f acc0 = v8f_zero(), acc1 = v8f_zero(), acc2 = v8f_zero(), acc3 = v8f_zero();

  const __bf16* srcs[2] = { A + (size_t)row0 * K, X + (size_t)row0 * K };
  const __bf16* wts[2]  = { WlT, WrT };

#pragma unroll
  for (int s = 0; s < 2; ++s) {
    const __bf16* strip = srcs[s];
    const __bf16* wt    = wts[s];
    for (int k0 = 0; k0 < K; k0 += 32) {
      const v16bf a  = load_a_frag(strip, K, k0, lane);
      const v16bf b0 = load_b_frag(wt, K, col0 +  0, k0, lane);
      const v16bf b1 = load_b_frag(wt, K, col0 + 16, k0, lane);
      const v16bf b2 = load_b_frag(wt, K, col0 + 32, k0, lane);
      const v16bf b3 = load_b_frag(wt, K, col0 + 48, k0, lane);
      acc0 = __builtin_amdgcn_wmma_f32_16x16x32_bf16(false, a, false, b0, (short)0, acc0, false, false);
      acc1 = __builtin_amdgcn_wmma_f32_16x16x32_bf16(false, a, false, b1, (short)0, acc1, false, false);
      acc2 = __builtin_amdgcn_wmma_f32_16x16x32_bf16(false, a, false, b2, (short)0, acc2, false, false);
      acc3 = __builtin_amdgcn_wmma_f32_16x16x32_bf16(false, a, false, b3, (short)0, acc3, false, false);
    }
  }

  // C/D layout: VGPR r -> lanes 0-15: (M=r, N=lane), lanes 16-31: (M=8+r, N=lane-16)
  const int half = lane >> 4;
  const int nl   = lane & 15;
  const float bv0 = bias[col0 +  0 + nl];
  const float bv1 = bias[col0 + 16 + nl];
  const float bv2 = bias[col0 + 32 + nl];
  const float bv3 = bias[col0 + 48 + nl];
#pragma unroll
  for (int r = 0; r < 8; ++r) {
    const size_t row = (size_t)(row0 + half * 8 + r);
    float v0 = acc0[r] + bv0;
    float v1 = acc1[r] + bv1;
    float v2 = acc2[r] + bv2;
    float v3 = acc3[r] + bv3;
    if (do_tanh) { v0 = tanhf(v0); v1 = tanhf(v1); v2 = tanhf(v2); v3 = tanhf(v3); }
    __bf16* zb = Zb + row * H_N + col0 + nl;
    zb[0]  = (__bf16)v0;
    zb[16] = (__bf16)v1;
    zb[32] = (__bf16)v2;
    zb[48] = (__bf16)v3;
    if (Zf) {
      float* zf = Zf + row * H_N + col0 + nl;
      zf[0] = v0; zf[16] = v1; zf[32] = v2; zf[48] = v3;
    }
  }
}

// ---------------------------------------------------------------------------
// Fused edge decoder (bf16 embeddings):
//   z = [zd[rows[l]], zp[cols[l]]]  (K = 512)
//   h = relu(z @ W1 + b1)           (N = 256, kept in registers)
//   out[l] = dot(h, W2) + b2        (in-register + 16-lane xor reduction)
// Wave: 16 rows, full N=256 (16 accumulators). Block: 8 waves = 128 rows.
// ---------------------------------------------------------------------------
__global__ __launch_bounds__(256, 1)
void decoder_wmma(const __bf16* __restrict__ zd, const __bf16* __restrict__ zp,
                  const int* __restrict__ rows, const int* __restrict__ cols,
                  const __bf16* __restrict__ W1T /* [256,512] */,
                  const float* __restrict__ b1, const float* __restrict__ W2,
                  const float* __restrict__ b2, float* __restrict__ out, int L) {
  const int lane = threadIdx.x & 31;
  const int wave = threadIdx.x >> 5;
  const int l0   = blockIdx.x * 128 + wave * 16;
  if (l0 >= L) return;

  const int m  = lane & 15;
  const int li = l0 + m;
  const __bf16* prow_d = zd + (size_t)rows[li] * H_N;
  const __bf16* prow_p = zp + (size_t)cols[li] * H_N;
  const int kb = (lane >> 4) << 3;   // 0 or 8

  v8f acc[16];
#pragma unroll
  for (int t = 0; t < 16; ++t) acc[t] = v8f_zero();

  for (int k0 = 0; k0 < 512; k0 += 32) {
    const __bf16* p = (k0 < 256) ? (prow_d + k0 + kb) : (prow_p + (k0 - 256) + kb);
    const v8bf lo = *reinterpret_cast<const v8bf*>(p);
    const v8bf hi = *reinterpret_cast<const v8bf*>(p + 16);
    v16bf a;
#pragma unroll
    for (int i = 0; i < 8; ++i) {
      a[i]     = lo[i];
      a[i + 8] = hi[i];
    }
#pragma unroll
    for (int t = 0; t < 16; ++t) {
      const v16bf b = load_b_frag(W1T, 512, t * 16, k0, lane);
      acc[t] = __builtin_amdgcn_wmma_f32_16x16x32_bf16(false, a, false, b, (short)0, acc[t], false, false);
    }
  }

  // epilogue: relu + weighted column sum, reduce across the 16 lanes of each half
  float part[8];
#pragma unroll
  for (int r = 0; r < 8; ++r) part[r] = 0.0f;
  const int nl = lane & 15;
#pragma unroll
  for (int t = 0; t < 16; ++t) {
    const int n     = t * 16 + nl;
    const float b1v = b1[n];
    const float w2v = W2[n];
#pragma unroll
    for (int r = 0; r < 8; ++r) {
      float h = acc[t][r] + b1v;
      h = h > 0.0f ? h : 0.0f;
      part[r] += h * w2v;
    }
  }
#pragma unroll
  for (int off = 1; off < 16; off <<= 1) {
#pragma unroll
    for (int r = 0; r < 8; ++r) part[r] += __shfl_xor(part[r], off, 32);
  }
  if (nl == 0) {
    const int rbase = l0 + (lane >> 4) * 8;
#pragma unroll
    for (int r = 0; r < 8; ++r) out[rbase + r] = part[r] + b2[0];
  }
}

// ---------------------------------------------------------------------------
// host
// ---------------------------------------------------------------------------
extern "C" void kernel_launch(void* const* d_in, const int* in_sizes, int n_in,
                              void* d_out, int out_size, void* d_ws, size_t ws_size,
                              hipStream_t stream) {
  (void)in_sizes; (void)n_in; (void)out_size; (void)ws_size;

  const float* x_drug  = (const float*)d_in[0];
  const float* x_prot  = (const float*)d_in[1];
  const int*   ei_src  = (const int*)d_in[2];
  const int*   ei_dst  = (const int*)d_in[3];
  const int*   eli_row = (const int*)d_in[4];
  const int*   eli_col = (const int*)d_in[5];
  const float* Wl_in_dp  = (const float*)d_in[6];
  const float* bl_in_dp  = (const float*)d_in[7];
  const float* Wr_in_dp  = (const float*)d_in[8];
  const float* Wl_in_pd  = (const float*)d_in[9];
  const float* bl_in_pd  = (const float*)d_in[10];
  const float* Wr_in_pd  = (const float*)d_in[11];
  const float* Wl_med_dp = (const float*)d_in[12];
  const float* bl_med_dp = (const float*)d_in[13];
  const float* Wr_med_dp = (const float*)d_in[14];
  const float* Wl_med_pd = (const float*)d_in[15];
  const float* bl_med_pd = (const float*)d_in[16];
  const float* Wr_med_pd = (const float*)d_in[17];
  const float* Wl_out_dp = (const float*)d_in[18];
  const float* bl_out_dp = (const float*)d_in[19];
  const float* Wr_out_dp = (const float*)d_in[20];
  const float* Wl_out_pd = (const float*)d_in[21];
  const float* bl_out_pd = (const float*)d_in[22];
  const float* Wr_out_pd = (const float*)d_in[23];
  const float* W1 = (const float*)d_in[24];
  const float* b1 = (const float*)d_in[25];
  const float* W2 = (const float*)d_in[26];
  const float* b2 = (const float*)d_in[27];

  // ---- carve workspace (256B aligned) ----
  uintptr_t cur = (uintptr_t)d_ws;
  auto carve = [&](size_t bytes) -> void* {
    void* p = (void*)cur;
    cur += (bytes + 255) & ~(size_t)255;
    return p;
  };
  // bf16 activation path
  __bf16* aggb   = (__bf16*)carve((size_t)ND_N * H_N * 2);   // reused both dirs
  __bf16* xb_d0  = (__bf16*)carve((size_t)ND_N * 128 * 2);   // bf16 copy of x_drug
  __bf16* xb_p0  = (__bf16*)carve((size_t)NP_N * 128 * 2);   // bf16 copy of x_prot
  __bf16* xdb_a  = (__bf16*)carve((size_t)ND_N * H_N * 2);
  __bf16* xdb_b  = (__bf16*)carve((size_t)ND_N * H_N * 2);
  __bf16* xpb_a  = (__bf16*)carve((size_t)NP_N * H_N * 2);
  __bf16* xpb_b  = (__bf16*)carve((size_t)NP_N * H_N * 2);
  __bf16* zdb    = (__bf16*)carve((size_t)ND_N * H_N * 2);   // final bf16 embeddings
  __bf16* zpb    = (__bf16*)carve((size_t)NP_N * H_N * 2);
  // CSR adjacency (built per launch, reused by all 4 layers)
  int* deg_p    = (int*)carve((size_t)NP_N * sizeof(int));
  int* offs_p   = (int*)carve((size_t)(NP_N + 1) * sizeof(int));
  int* cursor_p = (int*)carve((size_t)NP_N * sizeof(int));
  int* vals_p   = (int*)carve((size_t)E_N * sizeof(int));
  int* deg_d    = (int*)carve((size_t)ND_N * sizeof(int));
  int* offs_d   = (int*)carve((size_t)(ND_N + 1) * sizeof(int));
  int* cursor_d = (int*)carve((size_t)ND_N * sizeof(int));
  int* vals_d   = (int*)carve((size_t)E_N * sizeof(int));
  // transposed bf16 weights
  __bf16* wT_in_dp_l  = (__bf16*)carve((size_t)128 * 256 * 2);
  __bf16* wT_in_dp_r  = (__bf16*)carve((size_t)128 * 256 * 2);
  __bf16* wT_in_pd_l  = (__bf16*)carve((size_t)128 * 256 * 2);
  __bf16* wT_in_pd_r  = (__bf16*)carve((size_t)128 * 256 * 2);
  __bf16* wT_med_dp_l = (__bf16*)carve((size_t)256 * 256 * 2);
  __bf16* wT_med_dp_r = (__bf16*)carve((size_t)256 * 256 * 2);
  __bf16* wT_med_pd_l = (__bf16*)carve((size_t)256 * 256 * 2);
  __bf16* wT_med_pd_r = (__bf16*)carve((size_t)256 * 256 * 2);
  __bf16* wT_out_dp_l = (__bf16*)carve((size_t)256 * 256 * 2);
  __bf16* wT_out_dp_r = (__bf16*)carve((size_t)256 * 256 * 2);
  __bf16* wT_out_pd_l = (__bf16*)carve((size_t)256 * 256 * 2);
  __bf16* wT_out_pd_r = (__bf16*)carve((size_t)256 * 256 * 2);
  __bf16* wT_dec      = (__bf16*)carve((size_t)512 * 256 * 2);

  const dim3 blk(256);
  auto blocks_for = [](int n) { return dim3((unsigned)((n + 255) / 256)); };

  // ---- build CSR for both directions ----
  zero_i32<<<blocks_for(NP_N), blk, 0, stream>>>(deg_p, NP_N);
  zero_i32<<<blocks_for(ND_N), blk, 0, stream>>>(deg_d, ND_N);
  hist_i32<<<blocks_for(E_N), blk, 0, stream>>>(ei_dst, deg_p, E_N);
  hist_i32<<<blocks_for(E_N), blk, 0, stream>>>(ei_src, deg_d, E_N);
  scan_excl<<<1, 1024, 0, stream>>>(deg_p, offs_p, NP_N);
  scan_excl<<<1, 1024, 0, stream>>>(deg_d, offs_d, ND_N);
  copy_i32<<<blocks_for(NP_N), blk, 0, stream>>>(offs_p, cursor_p, NP_N);
  copy_i32<<<blocks_for(ND_N), blk, 0, stream>>>(offs_d, cursor_d, ND_N);
  // dp: protein <- list of drug sources ; pd: drug <- list of protein sources
  csr_place<<<blocks_for(E_N), blk, 0, stream>>>(ei_dst, ei_src, cursor_p, vals_p, E_N);
  csr_place<<<blocks_for(E_N), blk, 0, stream>>>(ei_src, ei_dst, cursor_d, vals_d, E_N);

  // ---- bf16 copies of input features ----
  f32_to_bf16<<<blocks_for(ND_N * 128), blk, 0, stream>>>(x_drug, xb_d0, ND_N * 128);
  f32_to_bf16<<<blocks_for(NP_N * 128), blk, 0, stream>>>(x_prot, xb_p0, NP_N * 128);

  // ---- convert + transpose weights to bf16 ----
  auto convw = [&](const float* W, __bf16* WT, int K) {
    const int total = K * 256;
    convert_wT<<<blocks_for(total), blk, 0, stream>>>(W, WT, K, 256);
  };
  convw(Wl_in_dp,  wT_in_dp_l,  128);
  convw(Wr_in_dp,  wT_in_dp_r,  128);
  convw(Wl_in_pd,  wT_in_pd_l,  128);
  convw(Wr_in_pd,  wT_in_pd_r,  128);
  convw(Wl_med_dp, wT_med_dp_l, 256);
  convw(Wr_med_dp, wT_med_dp_r, 256);
  convw(Wl_med_pd, wT_med_pd_l, 256);
  convw(Wr_med_pd, wT_med_pd_r, 256);
  convw(Wl_out_dp, wT_out_dp_l, 256);
  convw(Wr_out_dp, wT_out_dp_r, 256);
  convw(Wl_out_pd, wT_out_pd_l, 256);
  convw(Wr_out_pd, wT_out_pd_r, 256);
  convw(W1,        wT_dec,      512);

  // ---- one hetero SAGE layer (gather-sum + fused WMMA GEMM, both dirs) ----
  auto sage_layer = [&](const __bf16* xdb_in, const __bf16* xpb_in, int K,
                        const __bf16* WlT_dp, const __bf16* WrT_dp, const float* bl_dp,
                        const __bf16* WlT_pd, const __bf16* WrT_pd, const float* bl_pd,
                        __bf16* xdb_out, __bf16* xpb_out,
                        float* zdf, float* zpf, int act) {
    // drug -> protein (updates protein states)
    gather_sum_bf16<<<dim3((NP_N + 7) / 8), blk, 0, stream>>>(xdb_in, offs_p, vals_p, aggb, NP_N, K);
    sage_gemm_wmma<<<dim3((NP_N + 127) / 128, 4), blk, 0, stream>>>(
        aggb, xpb_in, WlT_dp, WrT_dp, bl_dp, xpb_out, zpf, NP_N, K, act);
    // protein -> drug (updates drug states)
    gather_sum_bf16<<<dim3((ND_N + 7) / 8), blk, 0, stream>>>(xpb_in, offs_d, vals_d, aggb, ND_N, K);
    sage_gemm_wmma<<<dim3((ND_N + 127) / 128, 4), blk, 0, stream>>>(
        aggb, xdb_in, WlT_pd, WrT_pd, bl_pd, xdb_out, zdf, ND_N, K, act);
  };

  float* zd_final  = (float*)d_out;
  float* zp_final  = (float*)d_out + (size_t)ND_N * H_N;
  float* dec_final = (float*)d_out + (size_t)(ND_N + NP_N) * H_N;

  // conv_in + tanh
  sage_layer(xb_d0, xb_p0, 128,
             wT_in_dp_l, wT_in_dp_r, bl_in_dp,
             wT_in_pd_l, wT_in_pd_r, bl_in_pd,
             xdb_a, xpb_a, nullptr, nullptr, 1);
  // conv_med x2 (shared weights) + tanh
  sage_layer(xdb_a, xpb_a, 256,
             wT_med_dp_l, wT_med_dp_r, bl_med_dp,
             wT_med_pd_l, wT_med_pd_r, bl_med_pd,
             xdb_b, xpb_b, nullptr, nullptr, 1);
  sage_layer(xdb_b, xpb_b, 256,
             wT_med_dp_l, wT_med_dp_r, bl_med_dp,
             wT_med_pd_l, wT_med_pd_r, bl_med_pd,
             xdb_a, xpb_a, nullptr, nullptr, 1);
  // conv_out, no activation: fp32 into d_out + bf16 copies for the decoder
  sage_layer(xdb_a, xpb_a, 256,
             wT_out_dp_l, wT_out_dp_r, bl_out_dp,
             wT_out_pd_l, wT_out_pd_r, bl_out_pd,
             zdb, zpb, zd_final, zp_final, 0);

  // fused edge decoder on bf16 embeddings
  decoder_wmma<<<dim3((L_N + 127) / 128), blk, 0, stream>>>(
      zdb, zpb, eli_row, eli_col, wT_dec, b1, W2, b2, dec_final, L_N);
}
